// GFNO_3D_70660801954403
// MI455X (gfx1250) — matrix operand order repaired and checked by reading, not compiled, not run on hardware
//
#include <hip/hip_runtime.h>

typedef __attribute__((ext_vector_type(16))) __bf16 v16bf;
typedef __attribute__((ext_vector_type(8)))  __bf16 v8bf;
typedef __attribute__((ext_vector_type(2)))  __bf16 v2bf;
typedef __attribute__((ext_vector_type(8)))  float  v8f;

#define CIN    64
#define COUT   64
#define D_IN   32
#define H_IN   64
#define W_IN   64
#define D_OUT  30
#define H_OUT  62
#define W_OUT  62
#define NSLICE 54                       // K = 64*27 = 1728, in slices of 32
#define APACK_ELEMS (4 * NSLICE * 32 * 16)   // 110592 bf16 = 221 KB

// ---------------------------------------------------------------------------
// Kernel 1: symmetrize W (O,1,I,G,3,3,3) -> dense (64,64,3,3,3), K-order
// kglob = tap*64 + c, packed into the per-lane WMMA A-fragment layout:
// element (mt, s, lane, i) = Wmat[mt*16 + (lane&15)][s*32 + K],
// K = (i<8 ? i : i+8) + (lane>=16 ? 8 : 0)   (ISA 16-bit A 16x32 table).
// ---------------------------------------------------------------------------
__global__ void build_packed_weights(const float* __restrict__ W,
                                     __bf16* __restrict__ Apack) {
  int idx  = blockIdx.x * 256 + threadIdx.x;     // < APACK_ELEMS
  int i    = idx & 15;
  int lane = (idx >> 4) & 31;
  int blk  = idx >> 9;                            // mt*NSLICE + s
  int s    = blk % NSLICE;
  int mt   = blk / NSLICE;
  int oc   = mt * 16 + (lane & 15);
  int K    = (i < 8 ? i : i + 8) + ((lane >> 4) << 3);
  int kg   = s * 32 + K;                          // global K
  int tap  = kg >> 6;                             // (kd*3+kh)*3+kw
  int c    = kg & 63;                             // input channel
  int kd = tap / 9, kh = (tap / 3) % 3, kw = tap % 3;
  int o  = oc >> 2, kc = oc & 3;                  // oc = o*4 + group copy
  int ii = c >> 2,  g  = c & 3;                   // c  = i_in*4 + group
  int sg = (g - kc) & 3;
  int sd, sh;                                     // rot^kc in (kD,kH) plane
  if      (kc == 0) { sd = kd;     sh = kh;     }
  else if (kc == 1) { sd = kh;     sh = 2 - kd; }
  else if (kc == 2) { sd = 2 - kd; sh = 2 - kh; }
  else              { sd = 2 - kh; sh = kd;     }
  float v = W[((((o * 16 + ii) * 4 + sg) * 3 + sd) * 3 + sh) * 3 + kw];
  Apack[idx] = (__bf16)v;
}

// ---------------------------------------------------------------------------
// Kernel 2: implicit-GEMM conv. One block per (n, d, h-pair): computes
// y[n][0:64][d][h0:h0+2][0:62].  Per GEMM: M=oc(64), N=w(64, 62 valid),
// K=1728.  8 waves = 2 h-rows x 4 waves; each wave owns a 32x32 output tile
// (2 M-tiles x 2 N-tiles, 4 accumulators) -> 1 ds_load_b128 per WMMA.
// LDS: xs[dz][dy][w][c] bf16, c innermost (128B rows), 16B-granule XOR
// swizzle (byteoff ^= (w&7)<<4).  3*4*64 rows * 128 B = 98304 B.
// ---------------------------------------------------------------------------
__launch_bounds__(256)
__global__ void gconv3d_wmma(const float* __restrict__ x,
                             const __bf16* __restrict__ Apack,
                             const float* __restrict__ bias,
                             float* __restrict__ y) {
  extern __shared__ __align__(16) char smem[];
  const int h0 = (int)blockIdx.x * 2, d = blockIdx.y, n = blockIdx.z;
  const int tid = (int)threadIdx.x;

  // ---- stage x[n, 0:64, d:d+3, h0:h0+4, 0:64] -> LDS bf16 (coalesced in w)
  {
    const int w   = tid & 63;
    const int swz = (w & 7) << 4;
    const float* xb = x + ((long)n * CIN * D_IN + d) * (H_IN * W_IN)
                        + (long)h0 * W_IN + w;
    for (int r = tid >> 6; r < (CIN / 2) * 12; r += 4) { // r=(c/2)*12+dz*4+dy
      int cp = r / 12, rem = r - cp * 12;
      int dz = rem >> 2, dy = rem & 3;
      long base = (long)(2 * cp) * (D_IN * H_IN * W_IN)
                + dz * (H_IN * W_IN) + dy * W_IN;
      float v0 = xb[base];
      float v1 = xb[base + (D_IN * H_IN * W_IN)];
      v2bf p = { (__bf16)v0, (__bf16)v1 };
      int row = rem * 64 + w;                           // (dz*4+dy)*64 + w
      *(v2bf*)(smem + row * 128 + ((cp * 4) ^ swz)) = p;
    }
  }
  __syncthreads();

  const int lane  = tid & 31;
  const int wid   = tid >> 5;
  const int lmod  = lane & 15;
  const int lhalf = lane >> 4;
  const int row   = wid >> 2;                // which of the two h rows
  const int q     = wid & 3;
  const int mtB   = (q >> 1) << 1;           // M-tile pair base (0 or 2)
  const int ntB   = (q & 1) << 1;            // N-tile pair base (0 or 2)
  const int nwa   = (ntB << 4) + lmod;       // first output column (<=47)
  const int nwb   = nwa + 16;                // second output column (<=63)

  v8f acc00 = {}, acc01 = {}, acc10 = {}, acc11 = {};
  const __bf16* Ap0 = Apack + ((long)mtB * NSLICE * 32 + lane) * 16;
  const __bf16* Ap1 = Ap0 + (long)NSLICE * 32 * 16;
  const int coB = lhalf << 4;                // +8 chans for lanes 16..31

  for (int kd = 0; kd < 3; ++kd)
    for (int kh = 0; kh < 3; ++kh) {
      const char* prow = smem + ((kd * 4 + row + kh) * 64) * 128;
      for (int kw = 0; kw < 3; ++kw) {
        int ca = nwa + kw; if (ca > 63) ca = 63;   // cols 62/63 discarded
        int cb = nwb + kw; if (cb > 63) cb = 63;
        const char* pa = prow + ca * 128; const int za = (ca & 7) << 4;
        const char* pb = prow + cb * 128; const int zb = (cb & 7) << 4;
#pragma unroll
        for (int ch = 0; ch < 2; ++ch) {           // two 32-chan tap halves
          const int co = coB + (ch << 6);
          v16bf a0 = *(const v16bf*)Ap0; Ap0 += 512;   // 2x global b128 (L2)
          v16bf a1 = *(const v16bf*)Ap1; Ap1 += 512;
          union Ub { v16bf v; v8bf hx[2]; } b0, b1;
          b0.hx[0] = *(const v8bf*)(pa + ((co     ) ^ za));
          b0.hx[1] = *(const v8bf*)(pa + ((co + 32) ^ za));
          b1.hx[0] = *(const v8bf*)(pb + ((co     ) ^ zb));
          b1.hx[1] = *(const v8bf*)(pb + ((co + 32) ^ zb));
          acc00 = __builtin_amdgcn_wmma_f32_16x16x32_bf16(
                      false, a0, false, b0.v, (short)0, acc00, false, false);
          acc10 = __builtin_amdgcn_wmma_f32_16x16x32_bf16(
                      false, a1, false, b0.v, (short)0, acc10, false, false);
          acc01 = __builtin_amdgcn_wmma_f32_16x16x32_bf16(
                      false, a0, false, b1.v, (short)0, acc01, false, false);
          acc11 = __builtin_amdgcn_wmma_f32_16x16x32_bf16(
                      false, a1, false, b1.v, (short)0, acc11, false, false);
        }
      }
    }

  // ---- epilogue: bias (repeat x4 along channel) + store
  const int hh = h0 + row;
#pragma unroll
  for (int mi = 0; mi < 2; ++mi) {
    const int ocb = ((mtB + mi) << 4) + (lhalf << 3);
    float* yp = y + ((((long)n * COUT + ocb) * D_OUT + d) * H_OUT + hh) * W_OUT;
    v8f aa = mi ? acc10 : acc00;
    v8f ab = mi ? acc11 : acc01;
#pragma unroll
    for (int vv = 0; vv < 8; ++vv) {
      float bv = bias[(ocb + vv) >> 2];
      yp[(long)vv * (D_OUT * H_OUT * W_OUT) + nwa] = aa[vv] + bv;  // nwa<=47
      if (nwb < W_OUT)
        yp[(long)vv * (D_OUT * H_OUT * W_OUT) + nwb] = ab[vv] + bv;
    }
  }
}

extern "C" void kernel_launch(void* const* d_in, const int* in_sizes, int n_in,
                              void* d_out, int out_size, void* d_ws, size_t ws_size,
                              hipStream_t stream) {
  const float* x  = (const float*)d_in[0];
  const float* W  = (const float*)d_in[1];
  const float* B  = (const float*)d_in[2];
  float* y        = (float*)d_out;
  __bf16* Apack   = (__bf16*)d_ws;                 // 221184 B scratch

  build_packed_weights<<<APACK_ELEMS / 256, 256, 0, stream>>>(W, Apack);
  dim3 grid(H_OUT / 2, D_OUT, 4);
  gconv3d_wmma<<<grid, 256, 3 * 4 * 64 * 128, stream>>>(x, Apack, B, y);
}
